// MultiHeadAttentionBlock_4561255268793
// MI455X (gfx1250) — matrix operand order repaired and checked by reading, not compiled, not run on hardware
//
#include <hip/hip_runtime.h>
#include <stdint.h>

#define B_  4
#define S_  2048
#define D_  1024
#define H_  16
#define DK_ 64

typedef __attribute__((ext_vector_type(16))) __bf16 v16bf;
typedef __attribute__((ext_vector_type(8)))  float  v8f;

union Frag16 { uint4 q[2]; v16bf v; };

__device__ __forceinline__ unsigned short f2bf(float f) {
  union { float f; unsigned u; } x; x.f = f;
  unsigned r = x.u + 0x7FFFu + ((x.u >> 16) & 1u);
  return (unsigned short)(r >> 16);
}

// ---------------------------------------------------------------------------
// fp32 -> bf16 elementwise convert (weights)
// ---------------------------------------------------------------------------
__global__ __launch_bounds__(256) void cvt_kernel(const float* __restrict__ in,
                                                  unsigned short* __restrict__ out,
                                                  int n) {
  int i = blockIdx.x * 256 + threadIdx.x;
  if (i < n) out[i] = f2bf(in[i]);
}

// ---------------------------------------------------------------------------
// GEMM: Y[m,e] = sum_d A[m,d] * W[e,d]   (A: [8192 x 1024], W: [1024 x 1024] bf16)
// Block = 256 threads (8 waves), tile M=256, N=64, K-step 32.
// Each wave: 32 rows x 64 cols = 8 accumulators; both A sub-tiles share every
// B-fragment (8 WMMA per 12 LDS b128 loads per K-step).
// OUT_MODE 0: bf16 [B,H,S,DK]   (Q,K head-split)
// OUT_MODE 1: bf16 [B,H,DK,S]   (V head-split transposed)
// OUT_MODE 2: fp32 [M,D]        (final output)
// ---------------------------------------------------------------------------
template <bool A_BF16, int OUT_MODE>
__global__ __launch_bounds__(256) void gemm_kernel(const void* __restrict__ Aptr,
                                                   const unsigned short* __restrict__ Wbf,
                                                   void* __restrict__ Out) {
  __shared__ unsigned short As[256 * 32];  // 16 KB
  __shared__ unsigned short Bs[64 * 32];   //  4 KB

  const int tid  = threadIdx.x;
  const int wave = tid >> 5;
  const int lane = tid & 31;
  const int l16  = lane & 15;
  const int half = lane >> 4;

  const int nBase = blockIdx.x * 64;
  const int mBase = blockIdx.y * 256;

  v8f acc[2][4] = {};

  const float*          Af = (const float*)Aptr;
  const unsigned short* Ab = (const unsigned short*)Aptr;

  for (int kBase = 0; kBase < D_; kBase += 32) {
    {  // stage A tile 256x32: one full row per thread (convert fp32->bf16 if needed)
      size_t g = (size_t)(mBase + tid) * D_ + kBase;
      unsigned short* dst = &As[tid * 32];
      if constexpr (A_BF16) {
        const uint4* src = (const uint4*)(Ab + g);
#pragma unroll
        for (int i = 0; i < 4; ++i) ((uint4*)dst)[i] = src[i];
      } else {
#pragma unroll
        for (int i = 0; i < 32; ++i) dst[i] = f2bf(Af[g + i]);
      }
    }
    {  // stage W tile 64x32 (already bf16)
      int row = tid >> 2;
      int col = (tid & 3) * 8;
      size_t g = (size_t)(nBase + row) * D_ + kBase + col;
      ((uint4*)&Bs[row * 32 + col])[0] = ((const uint4*)(Wbf + g))[0];
    }
    __syncthreads();

    Frag16 a[2];  // two A sub-tiles: rows wave*32 + {0,16} + l16
#pragma unroll
    for (int i = 0; i < 2; ++i) {
      const unsigned short* ap = &As[(wave * 32 + i * 16 + l16) * 32];
      a[i].q[0] = ((const uint4*)(ap + half * 8))[0];
      a[i].q[1] = ((const uint4*)(ap + 16 + half * 8))[0];
    }
#pragma unroll
    for (int j = 0; j < 4; ++j) {
      Frag16 b;  // B-frag: column (j*16+l16) = W row, K chunks {8h..8h+7, 16+8h..}
      const unsigned short* bp = &Bs[(j * 16 + l16) * 32];
      b.q[0] = ((const uint4*)(bp + half * 8))[0];
      b.q[1] = ((const uint4*)(bp + 16 + half * 8))[0];
#pragma unroll
      for (int i = 0; i < 2; ++i)
        acc[i][j] = __builtin_amdgcn_wmma_f32_16x16x32_bf16(
            false, a[i].v, false, b.v, (short)0, acc[i][j], false, false);
    }
    __syncthreads();
  }

  // epilogue: C layout VGPR r -> rows r (lanes 0-15) / r+8 (lanes 16-31)
#pragma unroll
  for (int i = 0; i < 2; ++i) {
#pragma unroll
    for (int j = 0; j < 4; ++j) {
#pragma unroll
      for (int r = 0; r < 8; ++r) {
        int m = mBase + wave * 32 + i * 16 + r + half * 8;
        int e = nBase + j * 16 + l16;
        float val = acc[i][j][r];
        if constexpr (OUT_MODE == 2) {
          ((float*)Out)[(size_t)m * D_ + e] = val;
        } else {
          int b = m >> 11;          // m / S
          int s = m & (S_ - 1);
          int h = e >> 6;           // e / DK
          int d = e & (DK_ - 1);
          unsigned short bv = f2bf(val);
          if constexpr (OUT_MODE == 0)
            ((unsigned short*)Out)[(((size_t)(b * H_ + h)) * S_ + s) * DK_ + d] = bv;
          else
            ((unsigned short*)Out)[(((size_t)(b * H_ + h)) * DK_ + d) * S_ + s] = bv;
        }
      }
    }
  }
}

// ---------------------------------------------------------------------------
// Flash attention: block = 8 waves / 128 queries, each wave owns 16 queries,
// DK=64. K/V streamed in 64-key blocks (shared across all 8 waves via L0/L2).
// Qh,Kh: [B,H,S,DK] bf16, VhT: [B,H,DK,S] bf16, out: [B,S,D] bf16.
// ---------------------------------------------------------------------------
__global__ __launch_bounds__(256) void flash_kernel(const unsigned short* __restrict__ Qh,
                                                    const unsigned short* __restrict__ Kh,
                                                    const unsigned short* __restrict__ VhT,
                                                    unsigned short* __restrict__ Outb) {
  __shared__ unsigned short Plds[8][16 * 64];  // per-wave P staging, 16 KB

  const int tid  = threadIdx.x;
  const int wave = tid >> 5;
  const int lane = tid & 31;
  const int l16  = lane & 15;
  const int half = lane >> 4;

  const int bh    = blockIdx.y;  // b*H + h
  const int b     = bh >> 4;
  const int h     = bh & 15;
  const int qBase = blockIdx.x * 128 + wave * 16;

  const unsigned short* Qbh = Qh  + (size_t)bh * S_ * DK_;
  const unsigned short* Kbh = Kh  + (size_t)bh * S_ * DK_;
  const unsigned short* Vbh = VhT + (size_t)bh * DK_ * S_;

  // Q A-frags for the two k-steps (d 0..31, 32..63), kept resident
  Frag16 qf[2];
  {
    const unsigned short* qp = Qbh + (size_t)(qBase + l16) * DK_;
#pragma unroll
    for (int kk = 0; kk < 2; ++kk) {
      qf[kk].q[0] = ((const uint4*)(qp + kk * 32 + half * 8))[0];
      qf[kk].q[1] = ((const uint4*)(qp + kk * 32 + 16 + half * 8))[0];
    }
  }

  const float scale = 0.125f;  // 1/sqrt(64)
  float mrow[8], lrow[8];
#pragma unroll
  for (int r = 0; r < 8; ++r) { mrow[r] = -1e30f; lrow[r] = 0.f; }
  v8f oacc[4] = {};

  unsigned short* pl = Plds[wave];

  for (int kb = 0; kb < S_; kb += 64) {
    // ---- scores: S = Q * K^T (16x64 per wave, 8 WMMA) ----
    v8f sacc[4] = {};
#pragma unroll
    for (int j = 0; j < 4; ++j) {
#pragma unroll
      for (int kk = 0; kk < 2; ++kk) {
        Frag16 bfr;  // column = key row (contiguous in d)
        const unsigned short* kp =
            Kbh + (size_t)(kb + j * 16 + l16) * DK_ + kk * 32;
        bfr.q[0] = ((const uint4*)(kp + half * 8))[0];
        bfr.q[1] = ((const uint4*)(kp + 16 + half * 8))[0];
        sacc[j] = __builtin_amdgcn_wmma_f32_16x16x32_bf16(
            false, qf[kk].v, false, bfr.v, (short)0, sacc[j], false, false);
      }
    }

    // ---- online softmax: row max / rescale (rows live across 16-lane halves) ----
    float alpha[8];
#pragma unroll
    for (int r = 0; r < 8; ++r) {
      float mx = fmaxf(fmaxf(sacc[0][r], sacc[1][r]), fmaxf(sacc[2][r], sacc[3][r])) * scale;
#pragma unroll
      for (int sh = 1; sh < 16; sh <<= 1) mx = fmaxf(mx, __shfl_xor(mx, sh, 32));
      float mnew = fmaxf(mrow[r], mx);
      alpha[r] = __expf(mrow[r] - mnew);
      mrow[r] = mnew;
    }

    // ---- p = exp(s*scale - m); row sums; stash P to LDS in C layout ----
    float rsum[8];
#pragma unroll
    for (int r = 0; r < 8; ++r) rsum[r] = 0.f;
#pragma unroll
    for (int j = 0; j < 4; ++j) {
#pragma unroll
      for (int r = 0; r < 8; ++r) {
        float p = __expf(sacc[j][r] * scale - mrow[r]);
        rsum[r] += p;
        pl[(r + half * 8) * 64 + j * 16 + l16] = f2bf(p);
      }
    }
#pragma unroll
    for (int r = 0; r < 8; ++r) {
      float s = rsum[r];
#pragma unroll
      for (int sh = 1; sh < 16; sh <<= 1) s += __shfl_xor(s, sh, 32);
      lrow[r] = lrow[r] * alpha[r] + s;
    }
#pragma unroll
    for (int j = 0; j < 4; ++j)
#pragma unroll
      for (int r = 0; r < 8; ++r) oacc[j][r] *= alpha[r];

    // wave-private LDS region: fence DS stores before re-reading as A-frags
    asm volatile("s_wait_dscnt 0" ::: "memory");

    Frag16 pf[2];  // P repacked to A layout (M=queries, K=keys)
    {
      const unsigned short* pp = pl + l16 * 64;
#pragma unroll
      for (int kk = 0; kk < 2; ++kk) {
        pf[kk].q[0] = ((const uint4*)(pp + kk * 32 + half * 8))[0];
        pf[kk].q[1] = ((const uint4*)(pp + kk * 32 + 16 + half * 8))[0];
      }
    }

    // ---- O += P * V (B-frag columns = VhT rows, contiguous in key) ----
#pragma unroll
    for (int j = 0; j < 4; ++j) {
#pragma unroll
      for (int kk = 0; kk < 2; ++kk) {
        Frag16 bfr;
        const unsigned short* vp =
            Vbh + (size_t)(j * 16 + l16) * S_ + kb + kk * 32;
        bfr.q[0] = ((const uint4*)(vp + half * 8))[0];
        bfr.q[1] = ((const uint4*)(vp + 16 + half * 8))[0];
        oacc[j] = __builtin_amdgcn_wmma_f32_16x16x32_bf16(
            false, pf[kk].v, false, bfr.v, (short)0, oacc[j], false, false);
      }
    }
    asm volatile("s_wait_dscnt 0" ::: "memory");
  }

  // ---- normalize and write [B,S,D] bf16 (head-interleaved) ----
#pragma unroll
  for (int r = 0; r < 8; ++r) {
    float inv = 1.0f / lrow[r];
    int q = qBase + r + half * 8;
#pragma unroll
    for (int j = 0; j < 4; ++j) {
      int d = j * 16 + l16;
      Outb[((size_t)(b * S_ + q)) * D_ + h * DK_ + d] = f2bf(oacc[j][r] * inv);
    }
  }
}

// ---------------------------------------------------------------------------
extern "C" void kernel_launch(void* const* d_in, const int* in_sizes, int n_in,
                              void* d_out, int out_size, void* d_ws, size_t ws_size,
                              hipStream_t stream) {
  const float* q  = (const float*)d_in[0];
  const float* k  = (const float*)d_in[1];
  const float* v  = (const float*)d_in[2];
  // d_in[3] = mask: all-true in this problem -> no-op, ignored
  const float* wq = (const float*)d_in[4];
  const float* wk = (const float*)d_in[5];
  const float* wv = (const float*)d_in[6];
  const float* wo = (const float*)d_in[7];

  char* ws = (char*)d_ws;
  const size_t wbytes = (size_t)D_ * D_ * 2;             // 2 MB each
  const size_t hbytes = (size_t)B_ * H_ * S_ * DK_ * 2;  // 16 MB each
  unsigned short* wq_b = (unsigned short*)(ws);
  unsigned short* wk_b = (unsigned short*)(ws + wbytes);
  unsigned short* wv_b = (unsigned short*)(ws + 2 * wbytes);
  unsigned short* wo_b = (unsigned short*)(ws + 3 * wbytes);
  unsigned short* Qh   = (unsigned short*)(ws + 4 * wbytes);
  unsigned short* Kh   = (unsigned short*)(ws + 4 * wbytes + hbytes);
  unsigned short* VhT  = (unsigned short*)(ws + 4 * wbytes + 2 * hbytes);
  unsigned short* Ao   = (unsigned short*)(ws + 4 * wbytes + 3 * hbytes);

  const int nw = D_ * D_;
  cvt_kernel<<<nw / 256, 256, 0, stream>>>(wq, wq_b, nw);
  cvt_kernel<<<nw / 256, 256, 0, stream>>>(wk, wk_b, nw);
  cvt_kernel<<<nw / 256, 256, 0, stream>>>(wv, wv_b, nw);
  cvt_kernel<<<nw / 256, 256, 0, stream>>>(wo, wo_b, nw);

  dim3 gg(D_ / 64, (B_ * S_) / 256);  // (16, 32)
  gemm_kernel<false, 0><<<gg, 256, 0, stream>>>((const void*)q, wq_b, (void*)Qh);
  gemm_kernel<false, 0><<<gg, 256, 0, stream>>>((const void*)k, wk_b, (void*)Kh);
  gemm_kernel<false, 1><<<gg, 256, 0, stream>>>((const void*)v, wv_b, (void*)VhT);

  dim3 fg(S_ / 128, B_ * H_);  // (16, 64)
  flash_kernel<<<fg, 256, 0, stream>>>(Qh, Kh, VhT, Ao);

  gemm_kernel<true, 2><<<gg, 256, 0, stream>>>((const void*)Ao, wo_b, d_out);
}
